// MaskedCausalAttention_50809463112320
// MI455X (gfx1250) — matrix operand order repaired and checked by reading, not compiled, not run on hardware
//
#include <hip/hip_runtime.h>
#include <hip/hip_bf16.h>

// ---------------------------------------------------------------------------
// Causal multi-head attention forward for MI455X (gfx1250, wave32, WMMA).
//   out  = softmax(mask(QK^T/sqrt(D))) V Wo + bo      (f32, first in d_out)
//   wts  = softmax weights (B,N,T,T)                  (f32, second in d_out)
// All GEMMs use v_wmma_f32_16x16x32_f16 with f16 operands / f32 accum.
// Causality exploited twice: masked score blocks write final 0.0 directly
// (softmax never touches them), and the P@V K-loop stops at the diagonal.
// ---------------------------------------------------------------------------

typedef _Float16 h16;
typedef __attribute__((ext_vector_type(16))) _Float16 v16h;
typedef __attribute__((ext_vector_type(8)))  _Float16 v8h;
typedef __attribute__((ext_vector_type(8)))  float    v8f;
typedef __attribute__((ext_vector_type(4)))  int      v4i;

static constexpr int BB = 4;
static constexpr int TT = 2048;
static constexpr int CC = 1024;
static constexpr int HH = 1024;
static constexpr int NH = 16;
static constexpr int DD = 64;
static constexpr int KSTEP = 32;   // WMMA K per step
static constexpr int LDK   = 40;   // padded LDS row stride (halves): 80B ->
                                   // 16 lanes sweep all 64 banks, no conflicts

static constexpr size_t QKV_SZ = (size_t)BB * NH * TT * DD;   // halves
static constexpr size_t OUT_SZ = (size_t)BB * TT * CC;        // floats

// gfx1250 async global->LDS copy path (compile-guarded; falls back to
// synchronous vector loads if the builtins aren't declared).
#if defined(__has_builtin)
# if __has_builtin(__builtin_amdgcn_global_load_async_to_lds_b128) && \
     __has_builtin(__builtin_amdgcn_s_wait_asynccnt)
#  define USE_ASYNC_LDS 1
# endif
#endif

#ifdef USE_ASYNC_LDS
typedef __attribute__((address_space(1))) v4i gas_v4i;   // global int4*
typedef __attribute__((address_space(3))) v4i las_v4i;   // LDS int4*
#endif

__device__ __forceinline__ void async_fence() {
#ifdef USE_ASYNC_LDS
  __builtin_amdgcn_s_wait_asynccnt(0);
#endif
}

// ---------------- WMMA helpers (ISA 7.12.2 layouts) -------------------------

__device__ __forceinline__ v8f wmma32(v16h a, v16h b, v8f c) {
  return __builtin_amdgcn_wmma_f32_16x16x32_f16(false, a, false, b,
                                                (short)0, c, false, false);
}

// A fragment 16x32 f16 from LDS tile laid out [row][k], k contiguous.
// lane<16: row=lane,    K in {0..7} U {16..23}
// lane>=16: row=lane-16, K in {8..15} U {24..31}
__device__ __forceinline__ v16h frag_a_lds(const h16* s, int row0, int lane) {
  const int m = row0 + (lane & 15);
  const int h = lane >> 4;
  v8h lo = *(const v8h*)(s + m * LDK + h * 8);
  v8h hi = *(const v8h*)(s + m * LDK + 16 + h * 8);
  return __builtin_shufflevector(lo, hi, 0,1,2,3,4,5,6,7,8,9,10,11,12,13,14,15);
}

// B fragment 32x16 f16 from LDS tile stored TRANSPOSED: [n][k], k contiguous.
// lane<16: col=lane, K in {0..7} U {16..23};  lane>=16: col=lane-16, K in {8..15} U {24..31}
__device__ __forceinline__ v16h frag_b_lds(const h16* s, int col0, int lane) {
  const int n = col0 + (lane & 15);
  const int h = lane >> 4;
  v8h lo = *(const v8h*)(s + n * LDK + h * 8);
  v8h hi = *(const v8h*)(s + n * LDK + 16 + h * 8);
  return __builtin_shufflevector(lo, hi, 0,1,2,3,4,5,6,7,8,9,10,11,12,13,14,15);
}

// ---------------- tile loaders (256 threads) --------------------------------

// 128x32 f32 tile -> f16 LDS [row][k]
__device__ __forceinline__ void load_f32_tile(h16* dst, const float* src,
                                              int ld, int row0, int k0, int tid) {
  const int r = tid >> 1;
  const int part = tid & 1;
  const float* p = src + (size_t)(row0 + r) * ld + k0 + part * 16;
  float4 f0 = ((const float4*)p)[0];
  float4 f1 = ((const float4*)p)[1];
  float4 f2 = ((const float4*)p)[2];
  float4 f3 = ((const float4*)p)[3];
  v8h a = {(h16)f0.x,(h16)f0.y,(h16)f0.z,(h16)f0.w,
           (h16)f1.x,(h16)f1.y,(h16)f1.z,(h16)f1.w};
  v8h b = {(h16)f2.x,(h16)f2.y,(h16)f2.z,(h16)f2.w,
           (h16)f3.x,(h16)f3.y,(h16)f3.z,(h16)f3.w};
  *(v8h*)(dst + r * LDK + part * 16)     = a;
  *(v8h*)(dst + r * LDK + part * 16 + 8) = b;
  __builtin_prefetch(p + KSTEP, 0, 1);   // next K-step (global_prefetch_b8)
}

// 128x32 f16 tile -> f16 LDS [row][k]
__device__ __forceinline__ void load_f16_tile(h16* dst, const h16* src,
                                              int ld, int row0, int k0, int tid) {
  const int r = tid >> 1;
  const int part = tid & 1;
  const h16* p = src + (size_t)(row0 + r) * ld + k0 + part * 16;
  h16* q = dst + r * LDK + part * 16;
#ifdef USE_ASYNC_LDS
  __builtin_amdgcn_global_load_async_to_lds_b128(
      (gas_v4i*)p, (las_v4i*)q, 0, 0);
  __builtin_amdgcn_global_load_async_to_lds_b128(
      (gas_v4i*)(p + 8), (las_v4i*)(q + 8), 0, 0);
#else
  v8h a = *(const v8h*)p;
  v8h b = *(const v8h*)(p + 8);
  *(v8h*)q       = a;
  *(v8h*)(q + 8) = b;
#endif
  __builtin_prefetch(p + KSTEP, 0, 1);
}

// 32(k) x 128(n) f32 weight tile -> f16 LDS TRANSPOSED [n][k]
__device__ __forceinline__ void load_w_tile_T(h16* dst, const float* W,
                                              int ld, int k0, int n0, int tid) {
  const int k  = tid & 31;
  const int nc = tid >> 5;                 // 0..7, 16 columns each
  const float* p = W + (size_t)(k0 + k) * ld + n0 + nc * 16;
#pragma unroll
  for (int i = 0; i < 4; ++i) {
    float4 f = ((const float4*)p)[i];
    const int nb = nc * 16 + i * 4;
    dst[(nb + 0) * LDK + k] = (h16)f.x;
    dst[(nb + 1) * LDK + k] = (h16)f.y;
    dst[(nb + 2) * LDK + k] = (h16)f.z;
    dst[(nb + 3) * LDK + k] = (h16)f.w;
  }
  __builtin_prefetch(p + (size_t)KSTEP * ld, 0, 1);
}

// 32(k=t') x 64(n=d) f16 V tile -> f16 LDS TRANSPOSED [d][t']
__device__ __forceinline__ void load_v_tile_T(h16* dst, const h16* V,
                                              int k0, int tid) {
  const int k  = tid & 31;
  const int dc = tid >> 5;                 // 0..7, 8 d each
  const h16* p = V + (size_t)(k0 + k) * DD + dc * 8;
  v8h v = *(const v8h*)p;
#pragma unroll
  for (int e = 0; e < 8; ++e) dst[(dc * 8 + e) * LDK + k] = v[e];
  __builtin_prefetch(p + KSTEP * DD, 0, 1);
}

// ---------------- kernel 1: QKV projection ---------------------------------
// grid (HH/128, B*T/128, 3); 256 thr = 8 waves, each wave 32(M)x64(N).
__global__ __launch_bounds__(256) void qkv_gemm_kernel(
    const float* __restrict__ x,
    const float* __restrict__ Wq, const float* __restrict__ bq,
    const float* __restrict__ Wk, const float* __restrict__ bk,
    const float* __restrict__ Wv, const float* __restrict__ bv,
    h16* __restrict__ qkv) {
  __shared__ h16 sA[128 * LDK];
  __shared__ h16 sB[128 * LDK];
  const int tid  = threadIdx.x;
  const int lane = tid & 31, wave = tid >> 5;
  const int wm = wave & 3, wn = wave >> 2;
  const int m0 = blockIdx.y * 128;
  const int n0 = blockIdx.x * 128;
  const int which = blockIdx.z;
  const float* W    = which == 0 ? Wq : which == 1 ? Wk : Wv;
  const float* bias = which == 0 ? bq : which == 1 ? bk : bv;
  h16* dst = qkv + (size_t)which * QKV_SZ;

  v8f acc[2][4];
#pragma unroll
  for (int i = 0; i < 2; ++i)
#pragma unroll
    for (int j = 0; j < 4; ++j) acc[i][j] = (v8f)(0.0f);

  for (int kt = 0; kt < CC; kt += KSTEP) {
    load_f32_tile(sA, x, CC, m0, kt, tid);
    load_w_tile_T(sB, W, HH, kt, n0, tid);
    __syncthreads();
    v16h a[2], b[4];
    a[0] = frag_a_lds(sA, wm * 32, lane);
    a[1] = frag_a_lds(sA, wm * 32 + 16, lane);
#pragma unroll
    for (int j = 0; j < 4; ++j) b[j] = frag_b_lds(sB, wn * 64 + j * 16, lane);
#pragma unroll
    for (int i = 0; i < 2; ++i)
#pragma unroll
      for (int j = 0; j < 4; ++j) acc[i][j] = wmma32(a[i], b[j], acc[i][j]);
    __syncthreads();
  }

  const int rsub = 8 * (lane >> 4);
#pragma unroll
  for (int i = 0; i < 2; ++i)
#pragma unroll
    for (int j = 0; j < 4; ++j) {
      const int n = n0 + wn * 64 + j * 16 + (lane & 15);
      const float bb = bias[n];
      const int head = n >> 6, d = n & (DD - 1);
#pragma unroll
      for (int e = 0; e < 8; ++e) {
        const int m = m0 + wm * 32 + i * 16 + rsub + e;
        const int bidx = m >> 11, t = m & (TT - 1);
        dst[((size_t)(bidx * NH + head) * TT + t) * DD + d] =
            (h16)(acc[i][j][e] + bb);
      }
    }
}

// ---------------- kernel 2: causal scores Q K^T ----------------------------
// grid (T/128 kcols, T/128 qrows, B*NH)
// Fully-masked tiles write the FINAL softmax value 0.0 (softmax skips them).
__global__ __launch_bounds__(256) void scores_kernel(
    const h16* __restrict__ Q, const h16* __restrict__ K,
    float* __restrict__ wts) {
  const int bz = blockIdx.z;
  const int q0 = blockIdx.y * 128;
  const int c0 = blockIdx.x * 128;
  float* out = wts + (size_t)bz * TT * TT;
  const int tid = threadIdx.x;

  if (c0 > q0 + 127) {             // fully masked tile -> final weight is 0
    for (int i = tid; i < 128 * 128; i += 256) {
      const int r = i >> 7, c = i & 127;
      out[(size_t)(q0 + r) * TT + c0 + c] = 0.0f;
    }
    return;
  }

  __shared__ h16 sQ[128 * LDK];
  __shared__ h16 sK[128 * LDK];
  const int lane = tid & 31, wave = tid >> 5;
  const int wm = wave & 3, wn = wave >> 2;
  const h16* q = Q + (size_t)bz * TT * DD;
  const h16* k = K + (size_t)bz * TT * DD;

  v8f acc[2][4];
#pragma unroll
  for (int i = 0; i < 2; ++i)
#pragma unroll
    for (int j = 0; j < 4; ++j) acc[i][j] = (v8f)(0.0f);

  for (int kt = 0; kt < DD; kt += KSTEP) {
    load_f16_tile(sQ, q, DD, q0, kt, tid);
    load_f16_tile(sK, k, DD, c0, kt, tid);   // K rows ARE B^T rows ([t'][d])
    async_fence();
    __syncthreads();
    v16h a[2], b[4];
    a[0] = frag_a_lds(sQ, wm * 32, lane);
    a[1] = frag_a_lds(sQ, wm * 32 + 16, lane);
#pragma unroll
    for (int j = 0; j < 4; ++j) b[j] = frag_b_lds(sK, wn * 64 + j * 16, lane);
#pragma unroll
    for (int i = 0; i < 2; ++i)
#pragma unroll
      for (int j = 0; j < 4; ++j) acc[i][j] = wmma32(a[i], b[j], acc[i][j]);
    __syncthreads();
  }

  const float scale = 0.125f;      // 1/sqrt(64)
  const int rsub = 8 * (lane >> 4);
#pragma unroll
  for (int i = 0; i < 2; ++i)
#pragma unroll
    for (int j = 0; j < 4; ++j) {
      const int tk = c0 + wn * 64 + j * 16 + (lane & 15);
#pragma unroll
      for (int e = 0; e < 8; ++e) {
        const int tq = q0 + wm * 32 + i * 16 + rsub + e;
        float s = acc[i][j][e] * scale;
        if (tk > tq) s = -__builtin_inff();   // diagonal block interior mask
        out[(size_t)tq * TT + tk] = s;
      }
    }
}

// ---------------- kernel 3: row softmax (in place, causal-truncated) -------
// grid (B*NH*T), 256 threads. Row t only has t+1 live entries; process
// ceil128(t+1) columns, the rest already hold the final value 0.
__global__ __launch_bounds__(256) void softmax_kernel(float* __restrict__ wts) {
  __shared__ float red[256];
  const size_t row = blockIdx.x;
  const int t = (int)(row & (TT - 1));
  const int nvalid = ((t >> 7) + 1) << 7;     // 128..2048
  float* p = wts + row * (size_t)TT;
  const int tid = threadIdx.x;
  float vals[8];
  float mx = -__builtin_inff();
#pragma unroll
  for (int i = 0; i < 8; ++i) {
    const int idx = tid + i * 256;
    vals[i] = (idx < nvalid) ? p[idx] : -__builtin_inff();
    mx = fmaxf(mx, vals[i]);
  }
  red[tid] = mx; __syncthreads();
  for (int s = 128; s > 0; s >>= 1) {
    if (tid < s) red[tid] = fmaxf(red[tid], red[tid + s]);
    __syncthreads();
  }
  mx = red[0]; __syncthreads();
  float sum = 0.0f;
#pragma unroll
  for (int i = 0; i < 8; ++i) { vals[i] = __expf(vals[i] - mx); sum += vals[i]; }
  red[tid] = sum; __syncthreads();
  for (int s = 128; s > 0; s >>= 1) {
    if (tid < s) red[tid] += red[tid + s];
    __syncthreads();
  }
  const float inv = 1.0f / red[0];
#pragma unroll
  for (int i = 0; i < 8; ++i) {
    const int idx = tid + i * 256;
    if (idx < nvalid) p[idx] = vals[i] * inv;
  }
}

// ---------------- kernel 4: attention = P @ V ------------------------------
// grid (1, T/128, B*NH); 8 waves each own 16 rows x 64 cols (1x4 tiles).
// Causality: P columns > row block are exactly 0 -> truncate K loop.
__global__ __launch_bounds__(256) void attnv_kernel(
    const float* __restrict__ wts, const h16* __restrict__ V,
    h16* __restrict__ att) {
  __shared__ h16 sP[128 * LDK];
  __shared__ h16 sV[DD * LDK];
  const int bz = blockIdx.z;
  const int q0 = blockIdx.y * 128;
  const int tid = threadIdx.x;
  const int lane = tid & 31, wave = tid >> 5;
  const float* P = wts + (size_t)bz * TT * TT;
  const h16* v = V + (size_t)bz * TT * DD;

  v8f acc[4];
#pragma unroll
  for (int j = 0; j < 4; ++j) acc[j] = (v8f)(0.0f);

  const int kend = q0 + 128;       // beyond this, P == 0
  for (int kt = 0; kt < kend; kt += KSTEP) {
    load_f32_tile(sP, P, TT, q0, kt, tid);
    load_v_tile_T(sV, v, kt, tid);
    __syncthreads();
    v16h a = frag_a_lds(sP, wave * 16, lane);
#pragma unroll
    for (int j = 0; j < 4; ++j)
      acc[j] = wmma32(a, frag_b_lds(sV, j * 16, lane), acc[j]);
    __syncthreads();
  }

  const int b = bz >> 4, head = bz & 15;
  const int rsub = 8 * (lane >> 4);
#pragma unroll
  for (int j = 0; j < 4; ++j) {
    const int d = j * 16 + (lane & 15);
#pragma unroll
    for (int e = 0; e < 8; ++e) {
      const int t = q0 + wave * 16 + rsub + e;
      att[((size_t)(b * TT + t)) * CC + head * DD + d] = (h16)acc[j][e];
    }
  }
}

// ---------------- kernel 5: output projection ------------------------------
// grid (HH/128, B*T/128)
__global__ __launch_bounds__(256) void outproj_kernel(
    const h16* __restrict__ att, const float* __restrict__ Wo,
    const float* __restrict__ bo, float* __restrict__ out) {
  __shared__ h16 sA[128 * LDK];
  __shared__ h16 sB[128 * LDK];
  const int tid  = threadIdx.x;
  const int lane = tid & 31, wave = tid >> 5;
  const int wm = wave & 3, wn = wave >> 2;
  const int m0 = blockIdx.y * 128;
  const int n0 = blockIdx.x * 128;

  v8f acc[2][4];
#pragma unroll
  for (int i = 0; i < 2; ++i)
#pragma unroll
    for (int j = 0; j < 4; ++j) acc[i][j] = (v8f)(0.0f);

  for (int kt = 0; kt < CC; kt += KSTEP) {
    load_f16_tile(sA, att, CC, m0, kt, tid);
    load_w_tile_T(sB, Wo, HH, kt, n0, tid);
    async_fence();
    __syncthreads();
    v16h a[2], b[4];
    a[0] = frag_a_lds(sA, wm * 32, lane);
    a[1] = frag_a_lds(sA, wm * 32 + 16, lane);
#pragma unroll
    for (int j = 0; j < 4; ++j) b[j] = frag_b_lds(sB, wn * 64 + j * 16, lane);
#pragma unroll
    for (int i = 0; i < 2; ++i)
#pragma unroll
      for (int j = 0; j < 4; ++j) acc[i][j] = wmma32(a[i], b[j], acc[i][j]);
    __syncthreads();
  }

  const int rsub = 8 * (lane >> 4);
#pragma unroll
  for (int i = 0; i < 2; ++i)
#pragma unroll
    for (int j = 0; j < 4; ++j) {
      const int n = n0 + wn * 64 + j * 16 + (lane & 15);
      const float bb = bo[n];
#pragma unroll
      for (int e = 0; e < 8; ++e) {
        const int m = m0 + wm * 32 + i * 16 + rsub + e;
        out[(size_t)m * HH + n] = acc[i][j][e] + bb;
      }
    }
}

// ---------------------------------------------------------------------------

extern "C" void kernel_launch(void* const* d_in, const int* in_sizes, int n_in,
                              void* d_out, int out_size, void* d_ws, size_t ws_size,
                              hipStream_t stream) {
  (void)in_sizes; (void)n_in; (void)out_size; (void)ws_size;
  const float* x  = (const float*)d_in[0];
  const float* Wq = (const float*)d_in[1];
  const float* bq = (const float*)d_in[2];
  const float* Wk = (const float*)d_in[3];
  const float* bk = (const float*)d_in[4];
  const float* Wv = (const float*)d_in[5];
  const float* bv = (const float*)d_in[6];
  const float* Wo = (const float*)d_in[7];
  const float* bo = (const float*)d_in[8];

  float* out = (float*)d_out;                 // (B,T,C)
  float* wts = out + OUT_SZ;                  // (B,N,T,T)

  h16* ws16 = (h16*)d_ws;
  h16* Qp   = ws16;                           // [b][h][t][d] f16
  h16* Kp   = ws16 + QKV_SZ;
  h16* Vp   = ws16 + 2 * QKV_SZ;
  h16* attp = ws16 + 3 * QKV_SZ;              // [b][t][h*64+d] f16

  const dim3 blk(256);
  qkv_gemm_kernel<<<dim3(HH / 128, (BB * TT) / 128, 3), blk, 0, stream>>>(
      x, Wq, bq, Wk, bk, Wv, bv, ws16);
  scores_kernel<<<dim3(TT / 128, TT / 128, BB * NH), blk, 0, stream>>>(
      Qp, Kp, wts);
  softmax_kernel<<<dim3(BB * NH * TT), blk, 0, stream>>>(wts);
  attnv_kernel<<<dim3(1, TT / 128, BB * NH), blk, 0, stream>>>(wts, Vp, attp);
  outproj_kernel<<<dim3(HH / 128, (BB * TT) / 128), blk, 0, stream>>>(
      attp, Wo, bo, out);
}
